// Transformer_34694745817610
// MI455X (gfx1250) — compile-verified, matrix operand order checked by reading, not run on hardware
//
// CDNA5 (gfx1250) transformer decoder forward.
// - All GEMMs on v_wmma_f32_16x16x32_bf16 (bf16 in, f32 accum).
// - Per-layer weights pre-converted to transposed bf16 [N][K]; weight B-tiles
//   staged into LDS by the Tensor Data Mover (tensor_load_to_lds + s_wait_tensorcnt).
// - Activation tiles staged with float4 loads + RNE fp32->bf16 packing.
// - All launch shapes are exact tile multiples -> no bounds checks anywhere.
#include <hip/hip_runtime.h>

#define B_ 4
#define T_ 512
#define S_ 512
#define D_ 768
#define H_ 12
#define DK_ 64
#define DFF_ 3072
#define LAYERS_ 6

typedef long long i64;
typedef unsigned long long u64;
typedef __attribute__((ext_vector_type(8)))  float          v8f;
typedef __attribute__((ext_vector_type(16))) __bf16         bf16x16;
typedef __attribute__((ext_vector_type(4)))  unsigned short u16x4;
typedef __attribute__((ext_vector_type(8)))  unsigned short u16x8;
typedef __attribute__((ext_vector_type(16))) unsigned short u16x16;
typedef __attribute__((ext_vector_type(4)))  unsigned int   uintx4;
typedef __attribute__((ext_vector_type(4)))  int            intx4;
typedef __attribute__((ext_vector_type(8)))  int            intx8;

__device__ __forceinline__ unsigned short f2bf(float f) {
  unsigned u = __builtin_bit_cast(unsigned, f);
  unsigned r = u + 0x7FFFu + ((u >> 16) & 1u);   // round-to-nearest-even
  return (unsigned short)(r >> 16);
}

struct GemmP {
  const float* A;      // fp32 [batchA][M][K], row stride a_rs
  const void*  Bm;     // mode0: fp32 [K][N]; mode1: fp32 [N][K]; mode2: bf16 [N][K]
  float*       C;
  const float* bias;   // per-N bias (may be null)
  int M, N, K;
  int a_rs, b_rs, c_rs;
  int a_div; i64 a_sdiv;                           // a_off = (z/a_div)*a_sdiv
  int b_mod; i64 b_smod;                           // b_off = (z%b_mod)*b_smod (elements)
  int c_div; i64 c_sdiv; int c_mod; i64 c_smod;    // c_off = (z/c_div)*c_sdiv + (z%c_mod)*c_smod
  int bias_mod; int bias_smod;
  int b_mode;                                      // 0 k-major f32, 1 n-major f32, 2 n-major bf16 via TDM
  int relu;
  float alpha;
};

#define BKT 32
#define LDSTR 40   // u16 row stride: 64B data + 16B pad = 80B; 16B-aligned, breaks bank periodicity

template <int BM, int BN>
__global__ __launch_bounds__(256) void wmma_gemm(GemmP p) {
  __shared__ __align__(16) unsigned short lA[BM * LDSTR];
  __shared__ __align__(16) unsigned short lB[BN * LDSTR];
  constexpr int WN = (BN == 128) ? 2 : 1;     // waves along N
  static_assert(BM / (8 / WN) == 32 && BN / WN == 64, "wave tile must be 32x64");

  const int tid   = threadIdx.x;
  const int z     = blockIdx.z;
  const int mTile = blockIdx.y * BM;
  const int nTile = blockIdx.x * BN;

  const i64 a_off    = (i64)(z / p.a_div) * p.a_sdiv;
  const i64 b_off    = (i64)(z % p.b_mod) * p.b_smod;
  const i64 c_off    = (i64)(z / p.c_div) * p.c_sdiv + (i64)(z % p.c_mod) * p.c_smod;
  const i64 bias_off = (i64)(z % p.bias_mod) * p.bias_smod;

  const int wid = tid >> 5;
  const int lane = tid & 31;
  const int hl  = lane >> 4;
  const int l16 = lane & 15;
  const int wm  = (wid / WN) * 32;
  const int wn  = (wid % WN) * 64;

  v8f acc[2][4] = {};

  for (int k0 = 0; k0 < p.K; k0 += BKT) {
    __syncthreads();

    // ---- A tile (BM x 32) fp32 -> bf16, vectorized, no guards (BM | M) ----
    {
      const float* Ab = p.A + a_off + k0;
      #pragma unroll
      for (int i = 0; i < BM / 32; ++i) {
        int e = i * 256 + tid;              // float4 chunk id
        int m = e >> 3, kq = (e & 7) * 4;
        float4 f = *(const float4*)&Ab[(i64)(mTile + m) * p.a_rs + kq];
        u16x4 pk = { f2bf(f.x), f2bf(f.y), f2bf(f.z), f2bf(f.w) };
        *(u16x4*)&lA[m * LDSTR + kq] = pk;
        if (k0 + BKT < p.K)                 // prefetch next k-tile of A
          __builtin_prefetch((const void*)&Ab[(i64)(mTile + m) * p.a_rs + kq + BKT], 0, 1);
      }
    }

    // ---- B tile as [n][k] in LDS ----
    if (p.b_mode == 2) {
      // bf16 n-major weights: Tensor Data Mover, one issue from wave 0.
      if (wid == 0) {
        u64 ga = (u64)p.Bm + 2ull * (u64)(b_off + (i64)nTile * p.b_rs + k0);
        unsigned lds = (unsigned)(u64)(size_t)&lB[0];
        uintx4 g0;
        g0[0] = 1u;                                        // count=1, user descriptor
        g0[1] = lds;                                       // lds_addr
        g0[2] = (unsigned)(ga & 0xffffffffull);            // global_addr[31:0]
        g0[3] = (unsigned)((ga >> 32) & 0x01ffffffull) | (2u << 30);  // addr[56:32] | type=2
        intx8 g1;
        g1[0] = (1 << 16) | (1 << 20) | (3 << 22) | (3 << 25);  // 2B elems, pad: every 16dw add 4dw
        g1[1] = (p.K & 0xffff) << 16;                      // tensor_dim0[15:0]
        g1[2] = ((p.K >> 16) & 0xffff) | ((p.N & 0xffff) << 16);
        g1[3] = ((p.N >> 16) & 0xffff) | (BKT << 16);      // tile_dim0 = 32
        g1[4] = BN;                                        // tile_dim1 rows (tile_dim2 = 0)
        g1[5] = p.K;                                       // tensor_dim0_stride[31:0]
        g1[6] = 0;
        g1[7] = 0;
        intx4 gz = {0, 0, 0, 0};
        intx8 gz8 = {0, 0, 0, 0, 0, 0, 0, 0};
        __builtin_amdgcn_tensor_load_to_lds(g0, g1, gz, gz, gz8, 0);
        __builtin_amdgcn_s_wait_tensorcnt(0);
      }
    } else if (p.b_mode == 1) {
      const float* Bb = (const float*)p.Bm + b_off + k0;
      #pragma unroll
      for (int i = 0; i < BN / 32; ++i) {
        int e = i * 256 + tid;
        int n = e >> 3, kq = (e & 7) * 4;
        float4 f = *(const float4*)&Bb[(i64)(nTile + n) * p.b_rs + kq];
        u16x4 pk = { f2bf(f.x), f2bf(f.y), f2bf(f.z), f2bf(f.w) };
        *(u16x4*)&lB[n * LDSTR + kq] = pk;
      }
    } else {
      // fp32 k-major -> transpose into LDS
      const float* Bb = (const float*)p.Bm + b_off + nTile;
      #pragma unroll
      for (int i = 0; i < BN / 32; ++i) {
        int e = i * 256 + tid;
        int nc = e % (BN / 4), kk = e / (BN / 4);
        float4 f = *(const float4*)&Bb[(i64)(k0 + kk) * p.b_rs + nc * 4];
        lB[(nc * 4 + 0) * LDSTR + kk] = f2bf(f.x);
        lB[(nc * 4 + 1) * LDSTR + kk] = f2bf(f.y);
        lB[(nc * 4 + 2) * LDSTR + kk] = f2bf(f.z);
        lB[(nc * 4 + 3) * LDSTR + kk] = f2bf(f.w);
      }
    }
    __syncthreads();

    // ---- fragments per CDNA5 16-bit A layout: K chunks at hl*8 and 16+hl*8 ----
    bf16x16 afr[2], bfr[4];
    #pragma unroll
    for (int mi = 0; mi < 2; ++mi) {
      int row = wm + mi * 16 + l16;
      u16x8 lo = *(const u16x8*)&lA[row * LDSTR + hl * 8];
      u16x8 hi = *(const u16x8*)&lA[row * LDSTR + 16 + hl * 8];
      u16x16 t = __builtin_shufflevector(lo, hi, 0,1,2,3,4,5,6,7,8,9,10,11,12,13,14,15);
      afr[mi] = __builtin_bit_cast(bf16x16, t);
    }
    #pragma unroll
    for (int ni = 0; ni < 4; ++ni) {
      int row = wn + ni * 16 + l16;
      u16x8 lo = *(const u16x8*)&lB[row * LDSTR + hl * 8];
      u16x8 hi = *(const u16x8*)&lB[row * LDSTR + 16 + hl * 8];
      u16x16 t = __builtin_shufflevector(lo, hi, 0,1,2,3,4,5,6,7,8,9,10,11,12,13,14,15);
      bfr[ni] = __builtin_bit_cast(bf16x16, t);
    }
    #pragma unroll
    for (int mi = 0; mi < 2; ++mi)
      #pragma unroll
      for (int ni = 0; ni < 4; ++ni)
        acc[mi][ni] = __builtin_amdgcn_wmma_f32_16x16x32_bf16(
            false, afr[mi], false, bfr[ni], (short)0, acc[mi][ni], false, false);
  }

  // ---- epilogue (no guards: tiles divide M,N exactly) ----
  #pragma unroll
  for (int ni = 0; ni < 4; ++ni) {
    int gn = nTile + wn + ni * 16 + l16;
    float bz = p.bias ? p.bias[bias_off + gn] : 0.0f;
    #pragma unroll
    for (int mi = 0; mi < 2; ++mi) {
      #pragma unroll
      for (int r = 0; r < 8; ++r) {
        int gm = mTile + wm + mi * 16 + r + 8 * hl;
        float v = acc[mi][ni][r] * p.alpha + bz;
        if (p.relu) v = fmaxf(v, 0.0f);
        p.C[c_off + (i64)gm * p.c_rs + gn] = v;
      }
    }
  }
}

// weight convert+transpose: in fp32 [z][K][N] -> out bf16 [z][N][K]
__global__ __launch_bounds__(256) void conv_tr(const float* in, unsigned short* out, int K, int N) {
  i64 zo = (i64)blockIdx.z * K * N;
  int idx = blockIdx.x * 256 + threadIdx.x;
  if (idx < K * N) {
    int n = idx / K, k = idx - n * K;
    out[zo + idx] = f2bf(in[zo + (i64)k * N + n]);
  }
}

// scores[b,h,q,:]: scale, mask (causal j>q | padding), softmax, in place
__global__ __launch_bounds__(256) void softmax_mask(float* sc, const unsigned char* pad,
                                                    int Tq, int Sk, float scale, int Hn) {
  const int qrow = blockIdx.x;
  const int z = blockIdx.y;
  const int b = z / Hn;
  float* row = sc + ((i64)z * Tq + qrow) * Sk;
  const unsigned char* pm = pad ? pad + (i64)b * Sk : nullptr;
  const int tid = threadIdx.x;
  __shared__ float red[256];
  float loc[4];
  const int nIt = (Sk + 255) / 256;
  float mx = -3.4e38f;
  for (int it = 0; it < nIt; ++it) {
    int j = tid + it * 256;
    float v = -3.4e38f;
    if (j < Sk) {
      bool masked = (j > qrow) || (pm && pm[j]);
      v = masked ? -3.4e38f : row[j] * scale;
    }
    loc[it] = v;
    mx = fmaxf(mx, v);
  }
  red[tid] = mx; __syncthreads();
  for (int s = 128; s > 0; s >>= 1) { if (tid < s) red[tid] = fmaxf(red[tid], red[tid + s]); __syncthreads(); }
  mx = red[0]; __syncthreads();
  float sum = 0.f;
  for (int it = 0; it < nIt; ++it) {
    float e = (loc[it] <= -3.3e38f) ? 0.f : __expf(loc[it] - mx);
    loc[it] = e; sum += e;
  }
  red[tid] = sum; __syncthreads();
  for (int s = 128; s > 0; s >>= 1) { if (tid < s) red[tid] += red[tid + s]; __syncthreads(); }
  float inv = 1.0f / red[0];
  for (int it = 0; it < nIt; ++it) { int j = tid + it * 256; if (j < Sk) row[j] = loc[it] * inv; }
}

// out[row] = LayerNorm(ca*A + cb*B + cc*C) * g + bt
__global__ __launch_bounds__(256) void add_ln(const float* A, const float* Bv, const float* Cv,
                                              float ca, float cb, float cc,
                                              const float* g, const float* bt, float* out, int Dd) {
  const i64 row = blockIdx.x;
  const int tid = threadIdx.x;
  __shared__ float red[256];
  float loc[4];
  const int nIt = (Dd + 255) / 256;
  float s = 0.f;
  for (int it = 0; it < nIt; ++it) {
    int i = tid + it * 256;
    float v = 0.f;
    if (i < Dd) {
      v = ca * A[row * Dd + i];
      if (Bv) v += cb * Bv[row * Dd + i];
      if (Cv) v += cc * Cv[row * Dd + i];
    }
    loc[it] = v; s += v;
  }
  red[tid] = s; __syncthreads();
  for (int st = 128; st > 0; st >>= 1) { if (tid < st) red[tid] += red[tid + st]; __syncthreads(); }
  float mean = red[0] / Dd; __syncthreads();
  float vs = 0.f;
  for (int it = 0; it < nIt; ++it) { int i = tid + it * 256; if (i < Dd) { float d = loc[it] - mean; vs += d * d; } }
  red[tid] = vs; __syncthreads();
  for (int st = 128; st > 0; st >>= 1) { if (tid < st) red[tid] += red[tid + st]; __syncthreads(); }
  float rstd = rsqrtf(red[0] / Dd + 1e-5f);
  for (int it = 0; it < nIt; ++it) {
    int i = tid + it * 256;
    if (i < Dd) out[row * Dd + i] = (loc[it] - mean) * rstd * g[i] + bt[i];
  }
}

// ---------------- host orchestration ----------------
static void launch_gemm(hipStream_t st,
                        const float* A, int a_rs, int a_div, i64 a_sdiv,
                        const void* Bm, int b_rs, int b_mod, i64 b_smod, int b_mode,
                        float* C, int c_rs, int c_div, i64 c_sdiv, int c_mod, i64 c_smod,
                        const float* bias, int bias_mod, int bias_smod,
                        int M, int N, int K, int batches, int relu, float alpha) {
  GemmP p;
  p.A = A; p.Bm = Bm; p.C = C; p.bias = bias;
  p.M = M; p.N = N; p.K = K;
  p.a_rs = a_rs; p.b_rs = b_rs; p.c_rs = c_rs;
  p.a_div = a_div; p.a_sdiv = a_sdiv;
  p.b_mod = b_mod; p.b_smod = b_smod;
  p.c_div = c_div; p.c_sdiv = c_sdiv; p.c_mod = c_mod; p.c_smod = c_smod;
  p.bias_mod = bias_mod; p.bias_smod = bias_smod;
  p.b_mode = b_mode; p.relu = relu; p.alpha = alpha;
  if (N % 128 == 0)
    wmma_gemm<128, 128><<<dim3(N / 128, M / 128, batches), 256, 0, st>>>(p);
  else
    wmma_gemm<256, 64><<<dim3(N / 64, M / 256, batches), 256, 0, st>>>(p);
}

struct AttnW {
  const unsigned short *Wqt, *Wkt, *Wvt, *Woht, *Wot;   // bf16 transposed [.. ][N][K]
  const float *bq, *bk, *bv, *boh, *bo;
};

static void run_attn(hipStream_t st, const AttnW& a, const float* xq, const float* xkv,
                     int Tq, int Sk, const unsigned char* pad,
                     float* qb, float* kb, float* vb, float* sc, float* ctx,
                     float* cc, float* out) {
  const int BH = B_ * H_;
  const int BIG = 1 << 30;
  // q = xq @ Wq[h] + bq[h]  -> [B,H,Tq,64]   (B: bf16 [H][64][768] via TDM)
  launch_gemm(st, xq, D_, H_, (i64)Tq * D_,
              a.Wqt, D_, H_, (i64)DK_ * D_, 2,
              qb, DK_, 1, (i64)Tq * DK_, 1, 0,
              a.bq, H_, DK_, Tq, DK_, D_, BH, 0, 1.f);
  launch_gemm(st, xkv, D_, H_, (i64)Sk * D_,
              a.Wkt, D_, H_, (i64)DK_ * D_, 2,
              kb, DK_, 1, (i64)Sk * DK_, 1, 0,
              a.bk, H_, DK_, Sk, DK_, D_, BH, 0, 1.f);
  launch_gemm(st, xkv, D_, H_, (i64)Sk * D_,
              a.Wvt, D_, H_, (i64)DK_ * D_, 2,
              vb, DK_, 1, (i64)Sk * DK_, 1, 0,
              a.bv, H_, DK_, Sk, DK_, D_, BH, 0, 1.f);
  // scores = q @ k^T  (k fp32 [Sk][64] is n-major)
  launch_gemm(st, qb, DK_, 1, (i64)Tq * DK_,
              kb, DK_, BIG, (i64)Sk * DK_, 1,
              sc, Sk, 1, (i64)Tq * Sk, 1, 0,
              nullptr, 1, 0, Tq, Sk, DK_, BH, 0, 1.f);
  softmax_mask<<<dim3(Tq, BH), dim3(256), 0, st>>>(sc, pad, Tq, Sk, 0.125f, H_);
  // ctx = att @ v  (v fp32 [Sk][64] is k-major)
  launch_gemm(st, sc, Sk, 1, (i64)Tq * Sk,
              vb, DK_, BIG, (i64)Sk * DK_, 0,
              ctx, DK_, 1, (i64)Tq * DK_, 1, 0,
              nullptr, 1, 0, Tq, DK_, Sk, BH, 0, 1.f);
  // per-head out proj, written into concat layout [B,Tq,H*64]
  launch_gemm(st, ctx, DK_, 1, (i64)Tq * DK_,
              a.Woht, DK_, H_, (i64)DK_ * DK_, 2,
              cc, D_, H_, (i64)Tq * D_, H_, DK_,
              a.boh, H_, DK_, Tq, DK_, DK_, BH, 0, 1.f);
  // out = concat @ Wo + bo
  launch_gemm(st, cc, D_, 1, 0,
              a.Wot, D_, 1, 0, 2,
              out, D_, 1, 0, 1, 0,
              a.bo, 1, 0, B_ * Tq, D_, D_, 1, 0, 1.f);
}

static void conv_weight(hipStream_t st, const float* in, unsigned short* out, int K, int N, int batches) {
  int total = K * N;
  conv_tr<<<dim3((total + 255) / 256, 1, batches), 256, 0, st>>>(in, out, K, N);
}

extern "C" void kernel_launch(void* const* d_in, const int* in_sizes, int n_in,
                              void* d_out, int out_size, void* d_ws, size_t ws_size,
                              hipStream_t stream) {
  (void)in_sizes; (void)n_in; (void)out_size; (void)ws_size;
  const float* x0   = (const float*)d_in[0];
  const float* enc  = (const float*)d_in[1];
  const float* bert = (const float*)d_in[2];
  const unsigned char* srcm = (const unsigned char*)d_in[3];
  const unsigned char* tgtm = (const unsigned char*)d_in[4];

  // params flattened jax-pytree style (sorted dict keys); 40 leaves/layer:
  // 0:W1 1:W2 2:b1 3:b2 | 4..13 bd{Wk,Wo,Woh,Wq,Wv,bk,bo,boh,bq,bv}
  // | 14..23 ed{...} | 24..29 ln1.b,ln1.g,ln2.b,ln2.g,ln3.b,ln3.g | 30..39 sa{...}
  auto leaf = [&](int l, int off) { return (const float*)d_in[5 + l * 40 + off]; };

  char* w = (char*)d_ws;
  auto allocB = [&](size_t bytes) {
    void* p = (void*)w;
    w += ((bytes + 255) & ~(size_t)255);
    return p;
  };
  auto allocF = [&](size_t elems) { return (float*)allocB(elems * 4); };
  auto allocH = [&](size_t elems) { return (unsigned short*)allocB(elems * 2); };

  const size_t BTd = (size_t)B_ * T_ * D_;
  float* qb   = allocF((size_t)B_ * H_ * 512 * DK_);
  float* kb   = allocF((size_t)B_ * H_ * 512 * DK_);
  float* vb   = allocF((size_t)B_ * H_ * 512 * DK_);
  float* sc   = allocF((size_t)B_ * H_ * T_ * S_);
  float* ctx  = allocF((size_t)B_ * H_ * T_ * DK_);
  float* cc   = allocF(BTd);
  float* atmp = allocF(BTd);
  float* saln = allocF(BTd);
  float* edo  = allocF(BTd);
  float* bdo  = allocF(BTd);
  float* fus  = allocF(BTd);
  float* ffh  = allocF((size_t)B_ * T_ * DFF_);
  float* ffo  = allocF(BTd);
  float* xbuf = allocF(BTd);

  // per-layer bf16 transposed weights (reused across layers)
  const size_t PRJ = (size_t)H_ * DK_ * D_;   // Wq/Wk/Wv transposed per head
  const size_t OHW = (size_t)H_ * DK_ * DK_;
  const size_t WOW = (size_t)D_ * D_;
  unsigned short* wt[3][5];  // [sa/ed/bd][Wq,Wk,Wv,Woh,Wo]
  for (int s = 0; s < 3; ++s) {
    wt[s][0] = allocH(PRJ); wt[s][1] = allocH(PRJ); wt[s][2] = allocH(PRJ);
    wt[s][3] = allocH(OHW); wt[s][4] = allocH(WOW);
  }
  unsigned short* w1t = allocH((size_t)DFF_ * D_);
  unsigned short* w2t = allocH((size_t)D_ * DFF_);

  const float* X = x0;
  for (int l = 0; l < LAYERS_; ++l) {
    const int abase[3] = {30, 14, 4};   // sa, ed, bd
    AttnW aw[3];
    for (int s = 0; s < 3; ++s) {
      int b = abase[s];
      const float* Wk = leaf(l, b + 0); const float* Wo = leaf(l, b + 1);
      const float* Woh = leaf(l, b + 2); const float* Wq = leaf(l, b + 3);
      const float* Wv = leaf(l, b + 4);
      conv_weight(stream, Wq, wt[s][0], D_, DK_, H_);
      conv_weight(stream, Wk, wt[s][1], D_, DK_, H_);
      conv_weight(stream, Wv, wt[s][2], D_, DK_, H_);
      conv_weight(stream, Woh, wt[s][3], DK_, DK_, H_);
      conv_weight(stream, Wo, wt[s][4], D_, D_, 1);
      aw[s].Wqt = wt[s][0]; aw[s].Wkt = wt[s][1]; aw[s].Wvt = wt[s][2];
      aw[s].Woht = wt[s][3]; aw[s].Wot = wt[s][4];
      aw[s].bk = leaf(l, b + 5); aw[s].bo = leaf(l, b + 6); aw[s].boh = leaf(l, b + 7);
      aw[s].bq = leaf(l, b + 8); aw[s].bv = leaf(l, b + 9);
    }
    const float* W1 = leaf(l, 0); const float* W2 = leaf(l, 1);
    const float* b1 = leaf(l, 2); const float* b2 = leaf(l, 3);
    conv_weight(stream, W1, w1t, D_, DFF_, 1);
    conv_weight(stream, W2, w2t, DFF_, D_, 1);
    const float* ln1b = leaf(l, 24); const float* ln1g = leaf(l, 25);
    const float* ln2b = leaf(l, 26); const float* ln2g = leaf(l, 27);
    const float* ln3b = leaf(l, 28); const float* ln3g = leaf(l, 29);

    // self-attention (causal + tgt padding), residual, ln1
    run_attn(stream, aw[0], X, X, T_, T_, tgtm, qb, kb, vb, sc, ctx, cc, atmp);
    add_ln<<<dim3(B_ * T_), dim3(256), 0, stream>>>(
        atmp, X, nullptr, 1.f, 1.f, 0.f, ln1g, ln1b, saln, D_);

    // cross attentions (reference applies causal mask to these too)
    run_attn(stream, aw[1], saln, enc, T_, S_, srcm, qb, kb, vb, sc, ctx, cc, edo);
    run_attn(stream, aw[2], saln, bert, T_, S_, srcm, qb, kb, vb, sc, ctx, cc, bdo);
    add_ln<<<dim3(B_ * T_), dim3(256), 0, stream>>>(
        edo, bdo, saln, 0.5f, 0.5f, 1.f, ln2g, ln2b, fus, D_);

    // FFN: relu(fused @ W1 + b1) @ W2 + b2
    launch_gemm(stream, fus, D_, 1, 0, w1t, D_, 1, 0, 2,
                ffh, DFF_, 1, 0, 1, 0, b1, 1, 0,
                B_ * T_, DFF_, D_, 1, 1, 1.f);
    launch_gemm(stream, ffh, DFF_, 1, 0, w2t, DFF_, 1, 0, 2,
                ffo, D_, 1, 0, 1, 0, b2, 1, 0,
                B_ * T_, D_, DFF_, 1, 0, 1.f);

    float* xout = (l == LAYERS_ - 1) ? (float*)d_out : xbuf;
    add_ln<<<dim3(B_ * T_), dim3(256), 0, stream>>>(
        ffo, fus, nullptr, 1.f, 1.f, 0.f, ln3g, ln3b, xout, D_);
    X = xbuf;
  }
}
